// Model_21483426414931
// MI455X (gfx1250) — matrix operand order repaired and checked by reading, not compiled, not run on hardware
//
#include <hip/hip_runtime.h>

typedef __attribute__((ext_vector_type(16))) _Float16 v16h;
typedef __attribute__((ext_vector_type(8)))  _Float16 v8h;
typedef __attribute__((ext_vector_type(8)))  float    v8f;

#define B_TOTAL 8192
#define SEQ     512
#define H       256
#define P       96

#define BT      16                 // batch rows per workgroup
#define WAVES   16                 // waves per workgroup (wave32) -> 8 waves/SIMD in CU-mode, 256-VGPR budget
#define THREADS (WAVES * 32)       // 512
#define NTW     3                  // N-tiles per wave: 16*3 = 48 tiles = 768 cols

// padded LDS row strides (avoid bank conflicts; keep 16B alignment)
#define H16S 264                   // h (f16) row stride in halves
#define H32S 260                   // h (f32) row stride in floats
#define GHS  772                   // gh (f32) row stride in floats
#define XS   516                   // x tile (f32) row stride

#define SMEM_BYTES (BT*GHS*4 + BT*H16S*2 + BT*H32S*4 + BT*XS*4 + 7*H*4)   // = 114688 (112 KB)

__device__ __forceinline__ float fsigmoid(float x) {
    float e = __builtin_amdgcn_exp2f(-1.4426950408889634f * x);
    return 1.0f / (1.0f + e);
}
__device__ __forceinline__ float ftanh(float x) {
    float e = __builtin_amdgcn_exp2f(-2.8853900817779268f * x);
    return 2.0f / (1.0f + e) - 1.0f;
}

__global__ void __launch_bounds__(THREADS)
gru_wmma_kernel(const float* __restrict__ x,      // [B, S]
                const float* __restrict__ W_ih,   // [3H, 1]
                const float* __restrict__ W_hh,   // [3H, H] row-major
                const float* __restrict__ b_ih,   // [3H]
                const float* __restrict__ b_hh,   // [3H]
                const float* __restrict__ W_out,  // [P, H]
                const float* __restrict__ b_out,  // [P]
                float* __restrict__ out)          // [B, P]
{
    extern __shared__ char smem[];
    // gh FIRST so every accumulator scatter store has an immediate DS offset (< 64 KB)
    float*    gh  = (float*)smem;                              // [BT][GHS]  h @ W_hh^T per step
    _Float16* h16 = (_Float16*)(gh + BT * GHS);                // [BT][H16S] f16 h (WMMA A source)
    float*    h32 = (float*)((char*)h16 + BT * H16S * 2);      // [BT][H32S] f32 h (precision carry)
    float*    xt  = h32 + BT * H32S;                           // [BT][XS]   staged x tile
    float*    prm = xt  + BT * XS;                             // 7*H params
    float* wr  = prm;         float* wz  = prm +   H; float* wn  = prm + 2*H;
    float* cr  = prm + 3*H;   float* cz  = prm + 4*H;
    float* cni = prm + 5*H;   float* cnh = prm + 6*H;

    const int tid  = threadIdx.x;
    const int lane = tid & 31;
    const int wv   = tid >> 5;
    const int b0   = blockIdx.x * BT;

    // ---------------- one-time setup ----------------
    for (int j = tid; j < H; j += THREADS) {
        wr[j]  = W_ih[j];
        wz[j]  = W_ih[H + j];
        wn[j]  = W_ih[2*H + j];
        cr[j]  = b_ih[j]       + b_hh[j];
        cz[j]  = b_ih[H + j]   + b_hh[H + j];
        cni[j] = b_ih[2*H + j];
        cnh[j] = b_hh[2*H + j];
    }
    for (int i = tid; i < BT * H16S; i += THREADS) h16[i] = (_Float16)0.0f;
    for (int i = tid; i < BT * H32S; i += THREADS) h32[i] = 0.0f;
    { // stage this block's 16 rows of x into LDS (wave w stages row w, coalesced)
        const float* xg = x + (size_t)(b0 + wv) * SEQ;
        for (int i = lane; i < SEQ; i += 32) xt[wv * XS + i] = xg[i];
    }

    // ---------------- preload W_hh B-fragments into VGPRs (held for all 512 steps) ----
    // B-fragment (32Kx16N): lane n = lane&15; per lane 16 consecutive K halves of W_hh
    // row n_global at k0 + 16*(lane>>4)
    const int koff = (lane >> 4) * 16;
    v16h Bf[NTW][8];
#pragma unroll
    for (int t = 0; t < NTW; ++t) {
        const int ng = (wv * NTW + t) * 16 + (lane & 15);
        const float* src = W_hh + (size_t)ng * H + koff;
#pragma unroll
        for (int kt = 0; kt < 8; ++kt) {
            v8h lo = __builtin_convertvector(*(const v8f*)(src + kt * 32),     v8h);
            v8h hi = __builtin_convertvector(*(const v8f*)(src + kt * 32 + 8), v8h);
            Bf[t][kt] = __builtin_shufflevector(lo, hi, 0,1,2,3,4,5,6,7,8,9,10,11,12,13,14,15);
        }
    }

    // -------- loop-invariant pointers (hoisted to minimize per-step temps) --------
    // A-fragment: lanes 0-15 read K[k0..k0+15] of row m; lanes 16-31 K[k0+16..k0+31]
    const _Float16* abase = h16 + (lane & 15) * H16S + (lane >> 4) * 16;
    // C/D scatter: VGPR r -> row (r + 8*(lane>>4)), col = tile_n*16 + (lane&15)
    float* gstore = gh + (8 * (lane >> 4)) * GHS + (wv * NTW) * 16 + (lane & 15);
    // gate-math row pointers: wave wv owns batch row wv; lane owns cols [lane*8, lane*8+8)
    const float* grow = gh  + wv * GHS;
    float*       hrow = h32 + wv * H32S;
    _Float16*    frow = h16 + wv * H16S;
    const float* xrow = xt  + wv * XS;
    const int    c0   = lane * 8;

    __syncthreads();

    // ---------------- recurrent time loop ----------------
    for (int s = 0; s < SEQ; ++s) {
        v8f acc0 = {}, acc1 = {}, acc2 = {};
#pragma unroll
        for (int kt = 0; kt < 8; ++kt) {
            const _Float16* ap = abase + kt * 32;
            v8h alo = *(const v8h*)(ap);
            v8h ahi = *(const v8h*)(ap + 8);
            v16h a  = __builtin_shufflevector(alo, ahi, 0,1,2,3,4,5,6,7,8,9,10,11,12,13,14,15);
            acc0 = __builtin_amdgcn_wmma_f32_16x16x32_f16(false, a, false, Bf[0][kt], (short)0, acc0, false, false);
            acc1 = __builtin_amdgcn_wmma_f32_16x16x32_f16(false, a, false, Bf[1][kt], (short)0, acc1, false, false);
            acc2 = __builtin_amdgcn_wmma_f32_16x16x32_f16(false, a, false, Bf[2][kt], (short)0, acc2, false, false);
        }
#pragma unroll
        for (int r = 0; r < 8; ++r) {
            gstore[r * GHS]      = acc0[r];
            gstore[r * GHS + 16] = acc1[r];
            gstore[r * GHS + 32] = acc2[r];
        }
        __syncthreads();

        // gate math (unroll 4 to keep live range small; loads still b128-vectorizable)
        {
            const float xs = xrow[s];
#pragma unroll 4
            for (int i = 0; i < 8; ++i) {
                const int j = c0 + i;
                const float g_r = grow[j];
                const float g_z = grow[H + j];
                const float g_n = grow[2*H + j];
                const float r_  = fsigmoid(xs * wr[j] + cr[j] + g_r);
                const float z_  = fsigmoid(xs * wz[j] + cz[j] + g_z);
                const float n_  = ftanh(xs * wn[j] + cni[j] + r_ * (g_n + cnh[j]));
                const float h0  = hrow[j];
                const float hn  = z_ * (h0 - n_) + n_;   // (1-z)*n + z*h
                hrow[j] = hn;
                frow[j] = (_Float16)hn;
            }
        }
        __syncthreads();
    }

    // ---------------- output projection: out = h_final @ W_out^T + b_out ----------------
    for (int idx = tid; idx < BT * P; idx += THREADS) {
        const int row = idx / P;
        const int p   = idx % P;
        const float* wrow2 = W_out + (size_t)p * H;
        const float* hrow2 = h32 + row * H32S;
        float sum = b_out[p];
#pragma unroll 8
        for (int k = 0; k < H; ++k) sum += hrow2[k] * wrow2[k];
        out[(size_t)(b0 + row) * P + p] = sum;
    }
}

extern "C" void kernel_launch(void* const* d_in, const int* in_sizes, int n_in,
                              void* d_out, int out_size, void* d_ws, size_t ws_size,
                              hipStream_t stream) {
    const float* x     = (const float*)d_in[0];
    const float* W_ih  = (const float*)d_in[1];
    const float* W_hh  = (const float*)d_in[2];
    const float* b_ih  = (const float*)d_in[3];
    const float* b_hh  = (const float*)d_in[4];
    const float* W_out = (const float*)d_in[5];
    const float* b_out = (const float*)d_in[6];
    float* out = (float*)d_out;

    dim3 grid(B_TOTAL / BT);      // 512 workgroups
    dim3 block(THREADS);          // 512 threads = 16 wave32
    gru_wmma_kernel<<<grid, block, SMEM_BYTES, stream>>>(x, W_ih, W_hh, b_ih, b_hh, W_out, b_out, out);
}
